// RSNN_ALIF_47442208751988
// MI455X (gfx1250) — matrix operand order repaired and checked by reading, not compiled, not run on hardware
//
#include <hip/hip_runtime.h>

// ---------------- problem constants (from reference) ----------------
constexpr int   N_T   = 200;
constexpr int   NB    = 128;    // batch
constexpr int   NFEAT = 512;
constexpr int   NIN   = 1024;   // 2*NFEAT
constexpr int   NREC  = 2048;
constexpr int   NOUT  = 32;
constexpr int   NRO   = 48;     // padded readout cols: 32 (w_out) + 1 (w_li) + 15 zero
constexpr float ALPHA = 0.8f, KAPPA = 0.8f, RHO = 0.97f, BETA = 0.07f, THR = 0.6f;
constexpr float SCALE = 50.0f;
constexpr float DT = 1e-3f, TAU_MEM_INV = 100.0f, TAU_SYN_INV = 200.0f, V_TH = 1.0f;

// ---------------- kernel geometry ----------------
constexpr int NWG    = 64;                 // persistent workgroups
constexpr int TPB    = 256;                // 8 wave32 per WG
constexpr int NPERWG = NREC / NWG;         // 32 neurons per WG
constexpr int KTOT   = NREC + NIN;         // 3072: concatenated [Wrec | Win] K dim
constexpr int LDS_ROW_DW = KTOT / 2 + 5;   // 1541 dwords/row: odd-ish stride -> conflict-free B loads

typedef __attribute__((ext_vector_type(16))) _Float16 v16h;
typedef __attribute__((ext_vector_type(2)))  _Float16 h2;
typedef __attribute__((ext_vector_type(8)))  float    v8f;

// ---------------- WMMA helpers ----------------
__device__ inline v8f wmma_f16(v16h a, v16h b, v8f c) {
  // D = A(16x32 f16) x B(32x16 f16) + C(16x16 f32)
  return __builtin_amdgcn_wmma_f32_16x16x32_f16(
      /*neg_a=*/false, a, /*neg_b=*/false, b,
      /*c_mod=*/(short)0, c, /*reuse_a=*/false, /*reuse_b=*/false);
}

// A fragment (16x32, M x K): lane<16 holds M=lane, K = k0..k0+15 contiguous;
// lane>=16 holds M=lane-16, base shifted by 16 halfs (8 dwords). kdw already lane-adjusted.
__device__ inline v16h ldfragA(const unsigned* rowu, int kdw) {
  v16h f;
  const h2* p = (const h2*)(rowu + kdw);
  const h2* q = (const h2*)(rowu + kdw + 4);
#pragma unroll
  for (int i = 0; i < 4; ++i) { h2 v = p[i]; f[2*i] = v.x; f[2*i+1] = v.y; }
#pragma unroll
  for (int i = 0; i < 4; ++i) { h2 v = q[i]; f[8+2*i] = v.x; f[9+2*i] = v.y; }
  return f;
}

// B fragment (32x16, K x N): N = lane%16 selects the row of W[n][k]; lanes>=16 take K+16.
// 8 consecutive dwords per lane (kdw already lane-adjusted).
__device__ inline v16h ldfragB(const unsigned* rowu, int kdw) {
  v16h f;
  const h2* p = (const h2*)(rowu + kdw);
#pragma unroll
  for (int i = 0; i < 8; ++i) { h2 v = p[i]; f[2*i] = v.x; f[2*i+1] = v.y; }
  return f;
}

// ---------------- grid-wide barrier (persistent kernel) ----------------
__device__ inline void grid_sync(unsigned* cnt, unsigned* gen, unsigned target) {
  __threadfence();
  __syncthreads();
  if (threadIdx.x == 0) {
    unsigned prev = atomicAdd(cnt, 1u);
    if (prev == (unsigned)NWG - 1u) {
      atomicExch(cnt, 0u);
      __threadfence();
      atomicAdd(gen, 1u);
    } else {
      while (atomicAdd(gen, 0u) < target) { __builtin_amdgcn_s_sleep(2); }
    }
    __threadfence();
  }
  __syncthreads();
}

// ---------------- prep kernels ----------------
__global__ void init_kernel(unsigned* zbuf_u, unsigned* sync) {
  unsigned idx = blockIdx.x * blockDim.x + threadIdx.x;
  if (idx < (unsigned)(2 * NB * NREC / 2)) zbuf_u[idx] = 0u;   // both z buffers = 0
  if (idx == 0) { sync[0] = 0u; sync[1] = 0u; }
}

// ConstantCurrentLIF encoder for +/- rectified inputs -> f16 spike tensor [N_T][NB][NIN]
__global__ void encode_kernel(const float* __restrict__ x, _Float16* __restrict__ xs) {
  int idx = blockIdx.x * blockDim.x + threadIdx.x;       // NB*NFEAT threads
  int b = idx / NFEAT, f = idx % NFEAT;
  float xv = SCALE * x[(size_t)b * NFEAT + f];
  float Ip = fmaxf(xv, 0.f), In = fmaxf(-xv, 0.f);
  float vp = 0.f, vn = 0.f;
  for (int t = 0; t < N_T; ++t) {
    vp += DT * TAU_MEM_INV * (Ip - vp);
    float zp = (vp > V_TH) ? 1.f : 0.f;
    vp *= (1.f - zp);
    vn += DT * TAU_MEM_INV * (In - vn);
    float zn = (vn > V_TH) ? 1.f : 0.f;
    vn *= (1.f - zn);
    size_t base = ((size_t)t * NB + b) * NIN;
    xs[base + f]         = (_Float16)zp;
    xs[base + NFEAT + f] = (_Float16)zn;
  }
}

// Concatenated f16 weights Wcat[n][k] = k<NREC ? (k==n ? 0 : w_rec[n][k]) : w_in[n][k-NREC]
__global__ void wcat_kernel(const float* __restrict__ w_rec, const float* __restrict__ w_in,
                            _Float16* __restrict__ wcat) {
  size_t idx = (size_t)blockIdx.x * blockDim.x + threadIdx.x;
  if (idx >= (size_t)NREC * KTOT) return;
  int n = (int)(idx / KTOT), k = (int)(idx % KTOT);
  float w = (k < NREC) ? ((k == n) ? 0.f : w_rec[(size_t)n * NREC + k])
                       : w_in[(size_t)n * NIN + (k - NREC)];
  wcat[idx] = (_Float16)w;
}

// Padded readout Wro[48][2048]: rows 0..31 = w_out, row 32 = w_li, rows 33..47 = 0
__global__ void wro_kernel(const float* __restrict__ w_out, const float* __restrict__ w_li,
                           _Float16* __restrict__ wro) {
  size_t idx = (size_t)blockIdx.x * blockDim.x + threadIdx.x;
  if (idx >= (size_t)NRO * NREC) return;
  int n = (int)(idx / NREC), k = (int)(idx % NREC);
  float w = (n < NOUT) ? w_out[(size_t)n * NREC + k] : ((n == NOUT) ? w_li[k] : 0.f);
  wro[idx] = (_Float16)w;
}

// ---------------- main persistent RSNN kernel ----------------
__global__ void __launch_bounds__(TPB, 1)
rsnn_kernel(const _Float16* __restrict__ xs, const _Float16* __restrict__ wcat,
            const _Float16* __restrict__ wro, _Float16* __restrict__ zbuf,
            float* __restrict__ mvo_ws, float* __restrict__ mli_ws, unsigned* __restrict__ sync) {
  extern __shared__ unsigned smem_u[];   // [NPERWG][LDS_ROW_DW] f16 weight slice (197 KB of 320 KB WGP LDS)

  const int lane   = threadIdx.x & 31;
  const int wave   = threadIdx.x >> 5;       // 0..7 -> M-tile (batch rows 16*wave..)
  const int m0     = wave * 16;
  const int n_base = blockIdx.x * NPERWG;    // this WG's 32 recurrent neurons
  const int l15    = lane & 15;
  const int hiA    = (lane & 16) ? 8 : 0;    // A frag dword shift (K+16 halfs)
  const int hiB    = (lane & 16) ? 8 : 0;    // B frag dword shift (K+16 halfs)
  const int rOff   = (lane & 16) ? 8 : 0;    // C/D: lanes>=16 hold M = r+8

  const unsigned* wcat_u = (const unsigned*)wcat;
  const unsigned* wro_u  = (const unsigned*)wro;
  const unsigned* xs_u   = (const unsigned*)xs;

  // ---- stage this WG's [32 x 3072] f16 weight slice into LDS (once) ----
  for (int idx = threadIdx.x; idx < NPERWG * (KTOT / 2); idx += TPB) {
    int row = idx / (KTOT / 2), col = idx % (KTOT / 2);
    smem_u[row * LDS_ROW_DW + col] = wcat_u[(size_t)(n_base + row) * (KTOT / 2) + col];
  }
  __syncthreads();

  const unsigned* brow0 = smem_u + (size_t)(0 * 16 + l15) * LDS_ROW_DW;
  const unsigned* brow1 = smem_u + (size_t)(1 * 16 + l15) * LDS_ROW_DW;
  const unsigned* wr0   = wro_u + (size_t)(0 * 16 + l15) * (NREC / 2);
  const unsigned* wr1   = wro_u + (size_t)(1 * 16 + l15) * (NREC / 2);
  const unsigned* wr2   = wro_u + (size_t)(2 * 16 + l15) * (NREC / 2);

  const v8f z8 = {0.f, 0.f, 0.f, 0.f, 0.f, 0.f, 0.f, 0.f};
  // persistent per-tile state in VGPRs across the whole time loop
  v8f vA0 = z8, vA1 = z8, aA0 = z8, aA1 = z8, zR0 = z8, zR1 = z8;
  // WG0 readout state
  v8f vo0 = z8, vo1 = z8, mv0 = z8, mv1 = z8, liV = z8, liI = z8, mL = z8;

  for (int t = 0; t < N_T - 1; ++t) {
    const _Float16* zc = zbuf + (size_t)(t & 1) * NB * NREC;
    _Float16*       zn = zbuf + (size_t)((t + 1) & 1) * NB * NREC;
    const unsigned* zc_u = (const unsigned*)zc;
    const unsigned* zn_u = (const unsigned*)zn;

    const unsigned* zrow = zc_u + (size_t)(m0 + l15) * (NREC / 2);
    const unsigned* xrow = xs_u + ((size_t)t * NB + m0 + l15) * (NIN / 2);
    if (t + 1 < N_T - 1)
      __builtin_prefetch(xs_u + ((size_t)(t + 1) * NB + m0 + l15) * (NIN / 2), 0, 0);

    // ---- phase 1: v' = alpha*v + z@Wrec^T + x_t@Win^T - thr*z ; spike ----
    v8f c0 = z8, c1 = z8;
#pragma unroll 4
    for (int k0 = 0; k0 < NREC; k0 += 32) {           // recurrent part (weights in LDS)
      v16h a  = ldfragA(zrow, k0 / 2 + hiA);
      v16h b0 = ldfragB(brow0, k0 / 2 + hiB);
      v16h b1 = ldfragB(brow1, k0 / 2 + hiB);
      c0 = wmma_f16(a, b0, c0);
      c1 = wmma_f16(a, b1, c1);
    }
#pragma unroll 4
    for (int k0 = 0; k0 < NIN; k0 += 32) {            // input part
      v16h a  = ldfragA(xrow, k0 / 2 + hiA);
      v16h b0 = ldfragB(brow0, (NREC + k0) / 2 + hiB);
      v16h b1 = ldfragB(brow1, (NREC + k0) / 2 + hiB);
      c0 = wmma_f16(a, b0, c0);
      c1 = wmma_f16(a, b1, c1);
    }

#pragma unroll
    for (int r = 0; r < 8; ++r) {
      const int bi = m0 + rOff + r;
      {
        float v = ALPHA * vA0[r] + c0[r] - THR * zR0[r];
        float a = RHO * aA0[r] + zR0[r];
        float z = (v > THR + BETA * a) ? 1.0f : 0.0f;
        vA0[r] = v; aA0[r] = a; zR0[r] = z;
        zn[(size_t)bi * NREC + n_base + l15] = (_Float16)z;
      }
      {
        float v = ALPHA * vA1[r] + c1[r] - THR * zR1[r];
        float a = RHO * aA1[r] + zR1[r];
        float z = (v > THR + BETA * a) ? 1.0f : 0.0f;
        vA1[r] = v; aA1[r] = a; zR1[r] = z;
        zn[(size_t)bi * NREC + n_base + 16 + l15] = (_Float16)z;
      }
    }

    grid_sync(sync, sync + 1, (unsigned)(t + 1));     // all new spikes visible

    // ---- phase 2 (WG0): readout  z' @ [w_out | w_li | 0]^T, running maxes ----
    if (blockIdx.x == 0) {
      const unsigned* zrowN = zn_u + (size_t)(m0 + l15) * (NREC / 2);
      v8f r0 = z8, r1 = z8, r2 = z8;
#pragma unroll 4
      for (int k0 = 0; k0 < NREC; k0 += 32) {
        v16h a = ldfragA(zrowN, k0 / 2 + hiA);
        r0 = wmma_f16(a, ldfragB(wr0, k0 / 2 + hiB), r0);
        r1 = wmma_f16(a, ldfragB(wr1, k0 / 2 + hiB), r1);
        r2 = wmma_f16(a, ldfragB(wr2, k0 / 2 + hiB), r2);
      }
#pragma unroll
      for (int r = 0; r < 8; ++r) {
        vo0[r] = KAPPA * vo0[r] + r0[r];  mv0[r] = fmaxf(mv0[r], vo0[r]);
        vo1[r] = KAPPA * vo1[r] + r1[r];  mv1[r] = fmaxf(mv1[r], vo1[r]);
        float ij = liI[r] + r2[r];                       // col 0 of tile 2 == w_li
        liV[r] = liV[r] + DT * TAU_MEM_INV * (ij - liV[r]);
        liI[r] = ij - DT * TAU_SYN_INV * ij;
        mL[r]  = fmaxf(mL[r], liV[r]);
      }
    }
  }

  // ---- dump running maxes (WG0 owns all batch rows across its 8 waves) ----
  if (blockIdx.x == 0) {
#pragma unroll
    for (int r = 0; r < 8; ++r) {
      const int bi = m0 + rOff + r;
      mvo_ws[(size_t)bi * NOUT + l15]      = mv0[r];
      mvo_ws[(size_t)bi * NOUT + 16 + l15] = mv1[r];
      if (l15 == 0) mli_ws[bi] = mL[r];
    }
  }
}

// ---------------- finalize: softmax(max(vo,0)) + critic ----------------
__global__ void finalize_kernel(const float* __restrict__ mvo, const float* __restrict__ mli,
                                float* __restrict__ out) {
  int b = threadIdx.x;
  if (b >= NB) return;
  float vals[NOUT];
  float mx = 0.f;                                   // maxes already clamped >= 0
#pragma unroll
  for (int j = 0; j < NOUT; ++j) { vals[j] = mvo[(size_t)b * NOUT + j]; mx = fmaxf(mx, vals[j]); }
  float s = 0.f;
#pragma unroll
  for (int j = 0; j < NOUT; ++j) { float e = __expf(vals[j] - mx); vals[j] = e; s += e; }
  float inv = 1.f / s;
#pragma unroll
  for (int j = 0; j < NOUT; ++j) out[(size_t)b * NOUT + j] = vals[j] * inv;
  out[(size_t)NB * NOUT + b] = mli[b];
}

// ---------------- host entry ----------------
extern "C" void kernel_launch(void* const* d_in, const int* in_sizes, int n_in,
                              void* d_out, int out_size, void* d_ws, size_t ws_size,
                              hipStream_t stream) {
  const float* x     = (const float*)d_in[0];
  const float* w_in  = (const float*)d_in[1];
  const float* w_rec = (const float*)d_in[2];
  const float* w_out = (const float*)d_in[3];
  const float* w_li  = (const float*)d_in[4];

  char* ws = (char*)d_ws;
  size_t off = 0;
  auto take = [&](size_t bytes) -> char* {
    char* p = ws + off;
    off = (off + bytes + 255) & ~(size_t)255;
    return p;
  };
  _Float16* xs   = (_Float16*)take((size_t)N_T * NB * NIN * sizeof(_Float16));   // ~52.4 MB
  _Float16* wcat = (_Float16*)take((size_t)NREC * KTOT * sizeof(_Float16));      // ~12.6 MB
  _Float16* wro  = (_Float16*)take((size_t)NRO * NREC * sizeof(_Float16));
  _Float16* zbuf = (_Float16*)take((size_t)2 * NB * NREC * sizeof(_Float16));    // double buffer
  float*    mvo  = (float*)take((size_t)NB * NOUT * sizeof(float));
  float*    mli  = (float*)take((size_t)NB * sizeof(float));
  unsigned* sync = (unsigned*)take(256);

  init_kernel<<<(2 * NB * NREC / 2 + 255) / 256, 256, 0, stream>>>((unsigned*)zbuf, sync);
  encode_kernel<<<(NB * NFEAT) / 256, 256, 0, stream>>>(x, xs);
  wcat_kernel<<<(int)(((size_t)NREC * KTOT + 255) / 256), 256, 0, stream>>>(w_rec, w_in, wcat);
  wro_kernel<<<(int)(((size_t)NRO * NREC + 255) / 256), 256, 0, stream>>>(w_out, w_li, wro);

  size_t lds_bytes = (size_t)NPERWG * LDS_ROW_DW * sizeof(unsigned);             // 197,248 B < 320 KB
  rsnn_kernel<<<NWG, TPB, lds_bytes, stream>>>(xs, wcat, wro, zbuf, mvo, mli, sync);

  finalize_kernel<<<1, 128, 0, stream>>>(mvo, mli, (float*)d_out);
}
